// BlockRecurrentMamba3_40046275068365
// MI455X (gfx1250) — compile-verified, matrix operand order checked by reading, not run on hardware
//
#include <hip/hip_runtime.h>

// ---- problem dims ----
#define D_M    1024
#define D_I    2048
#define N_ST   16
#define RK_    64
#define KCONV  4
#define LYR_   15
#define BS_    64
#define B_     4
#define L_     2048
#define T1_    65          // block tokens + state token
#define TP_    80          // padded rows per batch (multiple of 16)
#define ROWS_  (B_*TP_)    // 320
#define MT_    (ROWS_/16)  // 20 row tiles (GEMM covers 2 per wave)

typedef __attribute__((ext_vector_type(16))) __bf16 v16bf;
typedef __attribute__((ext_vector_type(8)))  __bf16 v8bf;
typedef __attribute__((ext_vector_type(8)))  float  v8f;

static __device__ __forceinline__ unsigned short f32_to_bf16(float f) {
  unsigned int u = __float_as_uint(f);
  u += 0x7fffu + ((u >> 16) & 1u);          // round-to-nearest-even
  return (unsigned short)(u >> 16);
}

// ---------------------------------------------------------------------------
// Pack fp32 weight [layers][K][N] -> bf16 WMMA B-fragment layout:
// tiles of 32(K)x16(N); per tile, lane l (n=l%16, khalf=l/16) holds 16 bf16:
//   j=0..7  : B[kt*32 +      khalf*8 + j][nt*16+n]
//   j=8..15 : B[kt*32 + 16 + khalf*8 + (j-8)][nt*16+n]
// so the GEMM loads each fragment as two contiguous b128 loads.
// ---------------------------------------------------------------------------
__global__ void pack_w_kernel(const float* __restrict__ src,
                              unsigned short* __restrict__ dst,
                              int K, int N, long total) {
  long idx = (long)blockIdx.x * 256 + threadIdx.x;
  if (idx >= total) return;
  int lane = (int)(idx & 31);
  long tile = idx >> 5;
  int ntiles = N >> 4, ktiles = K >> 5;
  int nt = (int)(tile % ntiles);
  long r = tile / ntiles;
  int kt = (int)(r % ktiles);
  int layer = (int)(r / ktiles);
  int nloc = lane & 15, khalf = lane >> 4;
  const float* s = src + (size_t)layer * K * N;
  unsigned short* d = dst + (size_t)idx * 16;
#pragma unroll
  for (int j = 0; j < 16; ++j) {
    int part = j >> 3, e = j & 7;
    int k = kt * 32 + part * 16 + khalf * 8 + e;
    d[j] = f32_to_bf16(s[(size_t)k * N + nt * 16 + nloc]);
  }
}

// ---------------------------------------------------------------------------
// Generic bf16 WMMA GEMM: C[ROWS_ x N] = A[ROWS_ x K] * W + bias, act.
// grid = (MT_/2, N/32); 1 wave per workgroup; 32x32 output tile per wave
// (2 M-tiles x 2 N-tiles -> 4 accumulators, 2x fragment reuse each way).
// act: 0 = none, 1 = softplus (for dt projection).
// ---------------------------------------------------------------------------
__global__ void gemm_bf16_wmma(const unsigned short* __restrict__ A,
                               const unsigned short* __restrict__ Wp,
                               float* __restrict__ C,
                               const float* __restrict__ bias,
                               int K, int N, int act) {
  int lane = threadIdx.x;
  int m0 = blockIdx.x * 32;
  int n0 = blockIdx.y * 32;
  int lmod = lane & 15, lhalf = lane >> 4;
  int ntiles = N >> 4;
  const unsigned short* arow0 = A + (size_t)(m0 + lmod) * K + lhalf * 8;
  const unsigned short* arow1 = arow0 + (size_t)16 * K;
  const unsigned short* wbase = Wp + (((size_t)(n0 >> 4)) * 32 + lane) * 16;
  size_t ktStride = (size_t)ntiles * 512;   // elems per K-tile row of tiles
  v8f acc00 = {}, acc01 = {}, acc10 = {}, acc11 = {};
  for (int kk = 0; kk < K; kk += 32) {
    union { v16bf v; v8bf h[2]; } a0, a1, b0, b1;
    a0.h[0] = *(const v8bf*)(arow0 + kk);
    a0.h[1] = *(const v8bf*)(arow0 + kk + 16);
    a1.h[0] = *(const v8bf*)(arow1 + kk);
    a1.h[1] = *(const v8bf*)(arow1 + kk + 16);
    const unsigned short* w0 = wbase + (size_t)(kk >> 5) * ktStride;
    b0.h[0] = *(const v8bf*)(w0);
    b0.h[1] = *(const v8bf*)(w0 + 8);
    b1.h[0] = *(const v8bf*)(w0 + 512);
    b1.h[1] = *(const v8bf*)(w0 + 520);
    if (kk + 32 < K) __builtin_prefetch(w0 + ktStride, 0, 1);  // global_prefetch_b8
    acc00 = __builtin_amdgcn_wmma_f32_16x16x32_bf16(false, a0.v, false, b0.v,
                                                    (short)0, acc00, false, false);
    acc01 = __builtin_amdgcn_wmma_f32_16x16x32_bf16(false, a0.v, false, b1.v,
                                                    (short)0, acc01, false, false);
    acc10 = __builtin_amdgcn_wmma_f32_16x16x32_bf16(false, a1.v, false, b0.v,
                                                    (short)0, acc10, false, false);
    acc11 = __builtin_amdgcn_wmma_f32_16x16x32_bf16(false, a1.v, false, b1.v,
                                                    (short)0, acc11, false, false);
  }
#pragma unroll
  for (int r = 0; r < 8; ++r) {
    int row0 = m0 + r + lhalf * 8;
    int row1 = row0 + 16;
    int c0 = n0 + lmod, c1 = n0 + 16 + lmod;
    float v00 = acc00[r], v01 = acc01[r], v10 = acc10[r], v11 = acc11[r];
    if (bias) {
      float bc0 = bias[c0], bc1 = bias[c1];
      v00 += bc0; v01 += bc1; v10 += bc0; v11 += bc1;
    }
    if (act == 1) {
      v00 = (v00 > 20.f) ? v00 : logf(1.f + __expf(v00));
      v01 = (v01 > 20.f) ? v01 : logf(1.f + __expf(v01));
      v10 = (v10 > 20.f) ? v10 : logf(1.f + __expf(v10));
      v11 = (v11 > 20.f) ? v11 : logf(1.f + __expf(v11));
    }
    C[(size_t)row0 * N + c0] = v00;
    C[(size_t)row0 * N + c1] = v01;
    C[(size_t)row1 * N + c0] = v10;
    C[(size_t)row1 * N + c1] = v11;
  }
}

// ---------------------------------------------------------------------------
// RMSNorm of residual tokens + state token prepend -> bf16 xin [ROWS_ x D_M]
// ---------------------------------------------------------------------------
__global__ void rmsnorm_xin_kernel(const float* __restrict__ outbuf,
                                   const float* __restrict__ statesL,
                                   const float* __restrict__ norm_wL,
                                   unsigned short* __restrict__ xin, int blk) {
  __shared__ float red[256];
  int r = blockIdx.x;                 // 0..319
  int b = r / TP_, t = r % TP_;
  int tid = threadIdx.x;
  unsigned short* xr = xin + (size_t)r * D_M;
  if (t == 0) {                       // state token (not normed)
    for (int d = tid; d < D_M; d += 256)
      xr[d] = f32_to_bf16(statesL[(size_t)b * D_M + d]);
    return;
  }
  if (t >= T1_) {                     // pad rows -> zero
    for (int d = tid; d < D_M; d += 256) xr[d] = 0;
    return;
  }
  const float* src = outbuf + ((size_t)b * L_ + (size_t)blk * BS_ + (t - 1)) * D_M;
  float ss = 0.f;
  for (int d = tid; d < D_M; d += 256) { float v = src[d]; ss += v * v; }
  red[tid] = ss; __syncthreads();
  for (int s = 128; s > 0; s >>= 1) { if (tid < s) red[tid] += red[tid + s]; __syncthreads(); }
  float scale = rsqrtf(red[0] * (1.f / D_M) + 1e-6f);
  for (int d = tid; d < D_M; d += 256)
    xr[d] = f32_to_bf16(src[d] * scale * norm_wL[d]);
}

// ---------------------------------------------------------------------------
// Causal depthwise conv (K=4) over xi (= first half of xz) + SiLU -> u (f32+bf16)
// ---------------------------------------------------------------------------
__global__ void conv_silu_kernel(const float* __restrict__ xz,
                                 const float* __restrict__ cwL,
                                 const float* __restrict__ cbL,
                                 float* __restrict__ u,
                                 unsigned short* __restrict__ ubf) {
  int idx = blockIdx.x * 256 + threadIdx.x;
  if (idx >= B_ * T1_ * D_I) return;
  int e = idx % D_I;
  int t = (idx / D_I) % T1_;
  int b = idx / (D_I * T1_);
  float acc = cbL[e];
#pragma unroll
  for (int i = 0; i < KCONV; ++i) {
    int tt = t - (KCONV - 1) + i;
    if (tt >= 0) acc += xz[((size_t)(b * TP_ + tt)) * (2 * D_I) + e] * cwL[i * D_I + e];
  }
  float sv = acc / (1.f + __expf(-acc));       // SiLU
  size_t ro = (size_t)(b * TP_ + t) * D_I + e;
  u[ro] = sv;
  ubf[ro] = f32_to_bf16(sv);
}

// dtr (first 64 cols of dbc) -> bf16 GEMM input
__global__ void dtr_cvt_kernel(const float* __restrict__ dbc,
                               unsigned short* __restrict__ dtr) {
  int idx = blockIdx.x * 256 + threadIdx.x;
  if (idx >= B_ * T1_ * RK_) return;
  int rr = idx % RK_;
  int t = (idx / RK_) % T1_;
  int b = idx / (RK_ * T1_);
  dtr[(size_t)(b * TP_ + t) * RK_ + rr] =
      f32_to_bf16(dbc[(size_t)(b * TP_ + t) * 96 + rr]);
}

// ---------------------------------------------------------------------------
// Selective scan fused with D-skip and SiLU(z) gating -> bf16 g [ROWS_ x D_I]
// thread owns (b, e); h[16] in registers; B_t/C_t broadcast via LDS.
// ---------------------------------------------------------------------------
__global__ void scan_gate_kernel(const float* __restrict__ dt,
                                 const float* __restrict__ u,
                                 const float* __restrict__ dbc,
                                 const float* __restrict__ xz,
                                 const float* __restrict__ AlogL,
                                 const float* __restrict__ DskL,
                                 unsigned short* __restrict__ g) {
  __shared__ float sB[N_ST], sC[N_ST];
  int tid = threadIdx.x;
  int b = blockIdx.x >> 3;
  int e = ((blockIdx.x & 7) << 8) + tid;
  float a[N_ST], h[N_ST];
#pragma unroll
  for (int n = 0; n < N_ST; ++n) {
    a[n] = -__expf(AlogL[(size_t)e * N_ST + n]);
    h[n] = 0.f;
  }
  float dsk = DskL[e];
  for (int t = 0; t < T1_; ++t) {
    size_t ro = (size_t)(b * TP_ + t);
    if (tid < N_ST)            sB[tid]      = dbc[ro * 96 + 64 + tid];
    else if (tid < 2 * N_ST)   sC[tid - 16] = dbc[ro * 96 + 80 + (tid - 16)];
    __syncthreads();
    float dtv = dt[ro * D_I + e];
    float uv  = u[ro * D_I + e];
    float dtu = dtv * uv;
    float y = 0.f;
#pragma unroll
    for (int n = 0; n < N_ST; ++n) {
      h[n] = __expf(dtv * a[n]) * h[n] + dtu * sB[n];
      y += h[n] * sC[n];
    }
    float zv = xz[ro * (2 * D_I) + D_I + e];
    float gv = (y + uv * dsk) * (zv / (1.f + __expf(-zv)));
    g[ro * D_I + e] = f32_to_bf16(gv);
    __syncthreads();
  }
}

// residual add into the output stream + extract carried state (last token)
__global__ void residual_state_kernel(const float* __restrict__ lo,
                                      float* __restrict__ outbuf,
                                      float* __restrict__ statesL, int blk) {
  int idx = blockIdx.x * 256 + threadIdx.x;    // B_*BS_*D_M
  int d = idx % D_M;
  int t = (idx / D_M) % BS_;
  int b = idx / (D_M * BS_);
  float lov = lo[((size_t)b * TP_ + t + 1) * D_M + d];
  outbuf[((size_t)b * L_ + (size_t)blk * BS_ + t) * D_M + d] += lov;
  if (t == BS_ - 1) statesL[(size_t)b * D_M + d] = lov;  // row 64 = new state
}

__global__ void init_states_kernel(const float* __restrict__ init_st,
                                   float* __restrict__ states) {
  int idx = blockIdx.x * 256 + threadIdx.x;    // LYR_*B_*D_M
  int d = idx % D_M;
  int l = idx / (D_M * B_);
  states[idx] = init_st[(size_t)l * D_M + d];
}

// ---------------------------------------------------------------------------
extern "C" void kernel_launch(void* const* d_in, const int* in_sizes, int n_in,
                              void* d_out, int out_size, void* d_ws, size_t ws_size,
                              hipStream_t stream) {
  (void)in_sizes; (void)n_in; (void)out_size; (void)ws_size;
  const float* x       = (const float*)d_in[0];
  const float* norm_w  = (const float*)d_in[1];
  const float* in_w    = (const float*)d_in[2];
  const float* conv_w  = (const float*)d_in[3];
  const float* conv_b  = (const float*)d_in[4];
  const float* xdbc_w  = (const float*)d_in[5];
  const float* dt_w    = (const float*)d_in[6];
  const float* dt_b    = (const float*)d_in[7];
  const float* A_log   = (const float*)d_in[8];
  const float* D_skip  = (const float*)d_in[9];
  const float* out_w   = (const float*)d_in[10];
  const float* init_st = (const float*)d_in[11];
  float* outbuf = (float*)d_out;

  char* ws = (char*)d_ws;
  size_t off = 0;
  auto take = [&](size_t bytes) -> char* {
    size_t o = (off + 255) & ~(size_t)255;
    off = o + bytes;
    return ws + o;
  };

  // per-layer packed-tile element counts
  const size_t PT_IN   = (size_t)(D_M / 32) * (4 * D_I / 16) * 512;   // 1024x8192
  const size_t PT_XDBC = (size_t)(D_I / 32) * (96 / 16) * 512;        // 2048x96
  const size_t PT_DT   = (size_t)(RK_ / 32) * (D_I / 16) * 512;       // 64x2048
  const size_t PT_OUT  = (size_t)(D_I / 32) * (D_M / 16) * 512;       // 2048x1024

  unsigned short* Wp_in   = (unsigned short*)take(LYR_ * PT_IN * 2);
  unsigned short* Wp_xdbc = (unsigned short*)take(LYR_ * PT_XDBC * 2);
  unsigned short* Wp_dt   = (unsigned short*)take(LYR_ * PT_DT * 2);
  unsigned short* Wp_out  = (unsigned short*)take(LYR_ * PT_OUT * 2);
  unsigned short* xin  = (unsigned short*)take((size_t)ROWS_ * D_M * 2);
  float*          xz   = (float*)take((size_t)ROWS_ * 2 * D_I * 4);
  float*          u    = (float*)take((size_t)ROWS_ * D_I * 4);
  unsigned short* ubf  = (unsigned short*)take((size_t)ROWS_ * D_I * 2);
  float*          dbc  = (float*)take((size_t)ROWS_ * 96 * 4);
  unsigned short* dtr  = (unsigned short*)take((size_t)ROWS_ * RK_ * 2);
  float*          dtv  = (float*)take((size_t)ROWS_ * D_I * 4);
  unsigned short* gbuf = (unsigned short*)take((size_t)ROWS_ * D_I * 2);
  float*          lo   = (float*)take((size_t)ROWS_ * D_M * 4);
  float*          states = (float*)take((size_t)LYR_ * B_ * D_M * 4);

  // ---- one-time per call: pack weights to bf16 fragment layout ----
  auto pack = [&](const float* src, unsigned short* dst, int K, int N) {
    long total = (long)LYR_ * (K / 32) * (N / 16) * 32;
    int blocks = (int)((total + 255) / 256);
    pack_w_kernel<<<dim3(blocks), dim3(256), 0, stream>>>(src, dst, K, N, total);
  };
  pack(in_w,   Wp_in,   D_M, 4 * D_I);
  pack(xdbc_w, Wp_xdbc, D_I, 96);
  pack(dt_w,   Wp_dt,   RK_, D_I);
  pack(out_w,  Wp_out,  D_I, D_M);

  init_states_kernel<<<dim3(LYR_ * B_ * D_M / 256), dim3(256), 0, stream>>>(init_st, states);
  hipMemcpyAsync(outbuf, x, (size_t)B_ * L_ * D_M * sizeof(float),
                 hipMemcpyDeviceToDevice, stream);   // residual stream = x

  const int convBlocks = (B_ * T1_ * D_I + 255) / 256;
  const int dtrBlocks  = (B_ * T1_ * RK_ + 255) / 256;

  for (int blk = 0; blk < L_ / BS_; ++blk) {
    for (int layer = 0; layer < LYR_; ++layer) {
      const float* nwL  = norm_w + (size_t)layer * D_M;
      const float* cwL  = conv_w + (size_t)layer * KCONV * D_I;
      const float* cbL  = conv_b + (size_t)layer * D_I;
      const float* dtbL = dt_b   + (size_t)layer * D_I;
      const float* AlL  = A_log  + (size_t)layer * D_I * N_ST;
      const float* DsL  = D_skip + (size_t)layer * D_I;
      float* stL = states + (size_t)layer * B_ * D_M;

      rmsnorm_xin_kernel<<<dim3(ROWS_), dim3(256), 0, stream>>>(outbuf, stL, nwL, xin, blk);

      gemm_bf16_wmma<<<dim3(MT_ / 2, (4 * D_I) / 32), dim3(32), 0, stream>>>(
          xin, Wp_in + (size_t)layer * PT_IN, xz, nullptr, D_M, 4 * D_I, 0);

      conv_silu_kernel<<<dim3(convBlocks), dim3(256), 0, stream>>>(xz, cwL, cbL, u, ubf);

      gemm_bf16_wmma<<<dim3(MT_ / 2, 96 / 32), dim3(32), 0, stream>>>(
          ubf, Wp_xdbc + (size_t)layer * PT_XDBC, dbc, nullptr, D_I, 96, 0);

      dtr_cvt_kernel<<<dim3(dtrBlocks), dim3(256), 0, stream>>>(dbc, dtr);

      gemm_bf16_wmma<<<dim3(MT_ / 2, D_I / 32), dim3(32), 0, stream>>>(
          dtr, Wp_dt + (size_t)layer * PT_DT, dtv, dtbL, RK_, D_I, 1);  // +bias, softplus

      scan_gate_kernel<<<dim3(B_ * (D_I / 256)), dim3(256), 0, stream>>>(
          dtv, u, dbc, xz, AlL, DsL, gbuf);

      gemm_bf16_wmma<<<dim3(MT_ / 2, D_M / 32), dim3(32), 0, stream>>>(
          gbuf, Wp_out + (size_t)layer * PT_OUT, lo, nullptr, D_I, D_M, 0);

      residual_state_kernel<<<dim3(B_ * BS_ * D_M / 256), dim3(256), 0, stream>>>(
          lo, outbuf, stL, blk);
    }
  }
}
